// CombineGraph_31310311588020
// MI455X (gfx1250) — matrix operand order repaired and checked by reading, not compiled, not run on hardware
//
#include <hip/hip_runtime.h>

#define B_ 128
#define L_ 50
#define D_ 128
#define S_ 12
#define ALPHA_ 0.2f
#define NEG_INF_ -9e15f

typedef __attribute__((ext_vector_type(16))) _Float16 v16h;
typedef __attribute__((ext_vector_type(8)))  _Float16 v8h;
typedef __attribute__((ext_vector_type(8)))  float    v8f;

__device__ __forceinline__ v8f wmma_f16(v16h a, v16h b, v8f c) {
  // D = A(16x32 f16) * B(32x16 f16) + C(16x16 f32)
  return __builtin_amdgcn_wmma_f32_16x16x32_f16(false, a, false, b, (short)0, c,
                                                false, false);
}

__device__ __forceinline__ v16h cat8(v8h lo, v8h hi8) {
  return __builtin_shufflevector(lo, hi8, 0, 1, 2, 3, 4, 5, 6, 7, 8, 9, 10, 11,
                                 12, 13, 14, 15);
}

// Fragment element orders (wave32, 16x16x32 f16):
//  A: lane (hi=l>>4, ln=l&15) row = row0+ln; halves = two contiguous 8-runs
//     at k0+16*hi and k0+16*hi+8 (16 contiguous).
//  B: lane col = col0+ln; halves = 16 contiguous at k0+16*hi.
//  C: VGPR i -> row i+8*hi, col = ln.

// ---------------------------------------------------------------------------
// Weight pre-swizzle: f32 [KT*32, 128] -> fragment-major f16, so each lane
// loads its whole v16h B-fragment from 32 contiguous bytes.
// ---------------------------------------------------------------------------
__global__ void swz_kernel(const float* __restrict__ src,
                           _Float16* __restrict__ dst, int KT) {
  int e = blockIdx.x * blockDim.x + threadIdx.x;
  int total = KT * 8 * 32 * 16;
  if (e >= total) return;
  int el = e & 15;
  int lane = (e >> 4) & 31;
  int f = e >> 9;
  int kt = f >> 3, nt = f & 7;
  int hi = lane >> 4, ln = lane & 15;
  int k = kt * 32 + hi * 16 + el;
  int col = nt * 16 + ln;
  dst[e] = (_Float16)src[k * 128 + col];
}

// ---------------------------------------------------------------------------
// LocalAggregator: per-batch edge-typed GAT + attn@h, all via WMMA.
// One workgroup (128 threads = 4 wave32) per batch b.
// ---------------------------------------------------------------------------
__global__ __launch_bounds__(128) void local_agg_kernel(
    const int* __restrict__ inputs, const int* __restrict__ adj,
    const float* __restrict__ emb,
    const float* __restrict__ a0, const float* __restrict__ a1,
    const float* __restrict__ a2, const float* __restrict__ a3,
    float* __restrict__ h_out, float* __restrict__ hl_out) {
  __shared__ alignas(32) _Float16 hSh[64 * 128];   // 16 KB h (f16, row-major)
  __shared__ alignas(32) _Float16 hTS[128 * 64];   // 16 KB h^T (f16)
  __shared__ alignas(32) _Float16 attnH[64 * 64];  // 8 KB
  __shared__ alignas(32) _Float16 aSh[4 * 128];    // 1 KB a_k (f16)
  __shared__ float         logitsS[64 * 64];       // 16 KB
  __shared__ unsigned char adjS[64 * 64];          // 4 KB

  const int b = blockIdx.x, tid = threadIdx.x;
  const int lane = tid & 31, wave = tid >> 5;
  const int hi = (lane >> 4) & 1, ln = lane & 15;

  // ---- stage h (gather from emb) as f16 + f16 transpose; persist f32 h ----
  for (int e = tid; e < 64 * 128; e += 128) {
    int l = e >> 7, d = e & 127;
    float v = 0.f;
    if (l < L_) {
      v = emb[(long)inputs[b * L_ + l] * D_ + d];
      h_out[(b * L_ + l) * D_ + d] = v;
    }
    _Float16 hf = (_Float16)v;
    hSh[l * 128 + d] = hf;
    hTS[d * 64 + l] = hf;
  }
  for (int e = tid; e < 4 * 128; e += 128) {
    const float* ak = (e < 128) ? a0 : (e < 256) ? a1 : (e < 384) ? a2 : a3;
    aSh[e] = (_Float16)ak[e & 127];
  }
  for (int e = tid; e < 64 * 64; e += 128) {
    int r = e >> 6, c = e & 63;
    adjS[e] = (r < L_ && c < L_) ? (unsigned char)adj[(b * L_ + r) * L_ + c]
                                 : (unsigned char)0;
    logitsS[e] = NEG_INF_;
    attnH[e] = (_Float16)0.f;
  }
  __syncthreads();

  // ---- e_k = (h .* a_k) @ h^T ; select by adj; leaky-relu ----
  for (int t = wave; t < 16; t += 4) {  // (edge type k) x (mtile)
    int k = t >> 2, mt = t & 3;
    const int arow = mt * 16 + ln;
    v16h af[4];  // A-fragments for all 4 k-steps, built once, packed f16 math
#pragma unroll
    for (int kt = 0; kt < 4; ++kt) {
      const int k0 = kt * 32;
      v8h h1 = *(const v8h*)&hSh[arow * 128 + k0 + 16 * hi];
      v8h h2 = *(const v8h*)&hSh[arow * 128 + k0 + 16 * hi + 8];
      v8h w1 = *(const v8h*)&aSh[k * 128 + k0 + 16 * hi];
      v8h w2 = *(const v8h*)&aSh[k * 128 + k0 + 16 * hi + 8];
      af[kt] = cat8(h1 * w1, h2 * w2);  // v_pk_mul_f16
    }
    for (int nt = 0; nt < 4; ++nt) {
      const int bcol = nt * 16 + ln;
      v8f acc = {};
#pragma unroll
      for (int kt = 0; kt < 4; ++kt)
        acc = wmma_f16(af[kt],
                       *(const v16h*)&hSh[bcol * 128 + kt * 32 + hi * 16], acc);
#pragma unroll
      for (int i = 0; i < 8; ++i) {
        int r = mt * 16 + i + hi * 8, c = nt * 16 + ln;
        if (adjS[r * 64 + c] == k + 1) {  // padding is 0: never matches
          float v = acc[i];
          logitsS[r * 64 + c] = (v > 0.f) ? v : ALPHA_ * v;
        }
      }
    }
  }
  __syncthreads();

  // ---- row softmax over j<50 (all-NEG_INF row -> uniform, matches jax) ----
  if (tid < L_) {
    int r = tid;
    float mx = NEG_INF_;
    for (int j = 0; j < L_; ++j) mx = fmaxf(mx, logitsS[r * 64 + j]);
    float s = 0.f;
    for (int j = 0; j < L_; ++j) {
      float e = __expf(logitsS[r * 64 + j] - mx);
      logitsS[r * 64 + j] = e;
      s += e;
    }
    float inv = 1.f / s;
    for (int j = 0; j < L_; ++j)
      attnH[r * 64 + j] = (_Float16)(logitsS[r * 64 + j] * inv);
  }
  __syncthreads();

  // ---- h_local = attn @ h ----
  for (int t = wave; t < 32; t += 4) {  // 4 mtiles x 8 ntiles
    int mt = t >> 3, nt = t & 7;
    v8f acc = {};
    const int arow = mt * 16 + ln, bcol = nt * 16 + ln;
#pragma unroll
    for (int kt = 0; kt < 2; ++kt) {  // K = 64 (padded j)
      const int k0 = kt * 32;
      v8h r1 = *(const v8h*)&attnH[arow * 64 + k0 + 16 * hi];
      v8h r2 = *(const v8h*)&attnH[arow * 64 + k0 + 16 * hi + 8];
      // B(k=j, n=d) = h[j,d] = hT[d][j]: 16 contiguous halves
      v16h bf = *(const v16h*)&hTS[bcol * 64 + k0 + hi * 16];
      acc = wmma_f16(cat8(r1, r2), bf, acc);
    }
#pragma unroll
    for (int i = 0; i < 8; ++i) {
      int r = mt * 16 + i + hi * 8, c = nt * 16 + ln;
      if (r < L_) hl_out[(b * L_ + r) * D_ + c] = acc[i];
    }
  }
}

// ---------------------------------------------------------------------------
// sum_item = masked session mean of emb[item]
// ---------------------------------------------------------------------------
__global__ __launch_bounds__(128) void sum_item_kernel(
    const int* __restrict__ item, const int* __restrict__ mask,
    const float* __restrict__ emb, float* __restrict__ sum_item) {
  int b = blockIdx.x, d = threadIdx.x;
  float s = 0.f, ms = 0.f;
  for (int l = 0; l < L_; ++l) {
    float m = (float)mask[b * L_ + l];
    ms += m;
    s += m * emb[(long)item[b * L_ + l] * D_ + d];
  }
  sum_item[b * D_ + d] = s / ms;
}

// ---------------------------------------------------------------------------
// Neighbor index / weight precompute (multi-hop sampling)
// ---------------------------------------------------------------------------
__global__ void nbr1_kernel(const int* __restrict__ inputs,
                            const int* __restrict__ adj_all,
                            const float* __restrict__ num_w,
                            int* __restrict__ idx1, float* __restrict__ w1) {
  int e = blockIdx.x * blockDim.x + threadIdx.x;
  if (e >= B_ * L_ * S_) return;
  int s = e % S_;
  int bl = e / S_;
  int node = inputs[bl];
  idx1[e] = adj_all[(long)node * S_ + s];
  w1[e]   = num_w[(long)node * S_ + s];
}

__global__ void nbr2_kernel(const int* __restrict__ idx1,
                            const int* __restrict__ adj_all,
                            const float* __restrict__ num_w,
                            int* __restrict__ idx2, float* __restrict__ w2) {
  int e = blockIdx.x * blockDim.x + threadIdx.x;
  if (e >= B_ * L_ * S_ * S_) return;
  int s = e % S_;
  int bm = e / S_;
  int node = idx1[bm];
  idx2[e] = adj_all[(long)node * S_ + s];
  w2[e]   = num_w[(long)node * S_ + s];
}

// ---------------------------------------------------------------------------
// GlobalAggregator: one WG per (b, tile of 16 m's). Neighbor embeddings are
// gathered straight from the L2-resident emb table (no entity[] buffers).
// Dynamic LDS (128 KB): qH = sum_item.*nv (f16), nvH (f16), staged gw1 frags.
// ---------------------------------------------------------------------------
__global__ __launch_bounds__(128) void global_agg_kernel(
    const float* __restrict__ emb, const float* __restrict__ sum_item,
    const float* __restrict__ svD, const int* __restrict__ sv_idx, int sv_gather,
    const float* __restrict__ nvD, const int* __restrict__ nv_idx, int nv_gather,
    const float* __restrict__ nw,
    const _Float16* __restrict__ gw1s,  // 32 frags (KT=4)
    const float* __restrict__ gw1l,     // gw1[128,:] (nw row)
    const float* __restrict__ gw2,      // [128]
    const _Float16* __restrict__ gw3s,  // 64 frags (KT=8)
    float* __restrict__ out, int M) {
  extern __shared__ char dynRaw[];
  _Float16* qH  = (_Float16*)dynRaw;             // 192*128 (48 KB)
  _Float16* nvH = qH + 192 * 128;                // 192*128 (48 KB)
  _Float16* g1S = nvH + 192 * 128;               // 32*32*16 (32 KB)

  __shared__ alignas(32) _Float16 svaggH[16 * 256];  // 8 KB: [sv | nv_agg]
  __shared__ float sumS[128];
  __shared__ float nwS[192];
  __shared__ float logitsS[192];
  __shared__ float attnS[192];
  __shared__ float g2S[128];
  __shared__ float g1lS[128];
  __shared__ int   nvIdxS[192];
  __shared__ int   svIdxS[16];

  const int b = blockIdx.y, m0 = blockIdx.x * 16;
  const int tid = threadIdx.x, lane = tid & 31, wave = tid >> 5;
  const int hi = (lane >> 4) & 1, ln = lane & 15;
  const int mcount = min(16, M - m0);

  // ---- stage scalars + indices + gw1 fragments ----
  sumS[tid] = sum_item[b * D_ + tid];
  g2S[tid] = gw2[tid];
  g1lS[tid] = gw1l[tid];
  for (int e = tid; e < 2048; e += 128)  // 32 KB of gw1 fragments as int4
    ((int4*)g1S)[e] = ((const int4*)gw1s)[e];
  for (int r = tid; r < 192; r += 128) {
    int ml = r / S_, s = r % S_;
    float w = 0.f;
    int gi = 0;
    if (ml < mcount) {
      long row = (long)b * M * S_ + (m0 + ml) * S_ + s;
      w = nw[row];
      gi = nv_gather ? nv_idx[row] : 0;
    }
    nwS[r] = w;
    nvIdxS[r] = gi;
  }
  if (tid < 16)
    svIdxS[tid] = (sv_gather && tid < mcount)
                      ? sv_idx[(long)b * M + m0 + tid] : 0;
  __syncthreads();

  // ---- stage nv rows (f16) + q = sum_item.*nv (f16) + sv rows ----
  for (int e = tid; e < 192 * 32; e += 128) {  // 192 rows x 32 float4
    int r = e >> 5, q = e & 31;
    int ml = r / S_;
    float4 v = make_float4(0.f, 0.f, 0.f, 0.f);
    if (ml < mcount) {
      long row = (long)b * M * S_ + (m0 + ml) * S_ + (r % S_);
      const float* src =
          nv_gather ? (emb + (long)nvIdxS[r] * D_) : (nvD + row * D_);
      v = ((const float4*)src)[q];
    }
    int base = r * 128 + q * 4;
    nvH[base + 0] = (_Float16)v.x; nvH[base + 1] = (_Float16)v.y;
    nvH[base + 2] = (_Float16)v.z; nvH[base + 3] = (_Float16)v.w;
    int d0 = q * 4;
    qH[base + 0] = (_Float16)(v.x * sumS[d0 + 0]);
    qH[base + 1] = (_Float16)(v.y * sumS[d0 + 1]);
    qH[base + 2] = (_Float16)(v.z * sumS[d0 + 2]);
    qH[base + 3] = (_Float16)(v.w * sumS[d0 + 3]);
  }
  for (int e = tid; e < 16 * 32; e += 128) {  // sv: 16 rows x 32 float4
    int ml = e >> 5, q = e & 31;
    float4 v = make_float4(0.f, 0.f, 0.f, 0.f);
    if (ml < mcount) {
      long row = (long)b * M + m0 + ml;
      const float* src =
          sv_gather ? (emb + (long)svIdxS[ml] * D_) : (svD + row * D_);
      v = ((const float4*)src)[q];
    }
    int base = ml * 256 + q * 4;
    svaggH[base + 0] = (_Float16)v.x; svaggH[base + 1] = (_Float16)v.y;
    svaggH[base + 2] = (_Float16)v.z; svaggH[base + 3] = (_Float16)v.w;
  }
  __syncthreads();

  // ---- phase A: (q || nw) @ gw1 -> leaky -> . gw2 -> logits ----
  for (int rt = wave; rt < 12; rt += 4) {  // 12 row-tiles of 16 (m,s) rows
    const int arow = rt * 16 + ln;
    v16h af[4];  // A-fragments hoisted out of the nt loop: pure LDS copies
#pragma unroll
    for (int kt = 0; kt < 4; ++kt) {
      const int k0 = kt * 32;
      af[kt] = cat8(*(const v8h*)&qH[arow * 128 + k0 + 16 * hi],
                    *(const v8h*)&qH[arow * 128 + k0 + 16 * hi + 8]);
    }
    float p[8];
#pragma unroll
    for (int i = 0; i < 8; ++i) p[i] = 0.f;
    for (int nt = 0; nt < 8; ++nt) {
      v8f acc = {};
#pragma unroll
      for (int kt = 0; kt < 4; ++kt)
        acc = wmma_f16(af[kt],
                       *(const v16h*)&g1S[((kt * 8 + nt) * 32 + lane) * 16],
                       acc);
      int c = nt * 16 + ln;
      float g2 = g2S[c];
      float g1last = g1lS[c];
#pragma unroll
      for (int i = 0; i < 8; ++i) {
        int r = rt * 16 + i + hi * 8;
        float v = acc[i] + nwS[r] * g1last;
        v = (v > 0.f) ? v : 0.2f * v;  // leaky-relu(0.2)
        p[i] += v * g2;
      }
    }
    // reduce the 16 column-lanes of each half-wave (butterfly)
#pragma unroll
    for (int i = 0; i < 8; ++i) {
      float v = p[i];
      v += __shfl_xor(v, 1, 32);
      v += __shfl_xor(v, 2, 32);
      v += __shfl_xor(v, 4, 32);
      v += __shfl_xor(v, 8, 32);
      if (ln == 0) logitsS[rt * 16 + i + hi * 8] = v;  // T == 1.0
    }
  }
  __syncthreads();

  // ---- phase B: softmax over S=12 per m ----
  if (tid < 16) {
    int m = tid;
    float mx = -1e30f;
    for (int s = 0; s < S_; ++s) mx = fmaxf(mx, logitsS[m * S_ + s]);
    float sm = 0.f;
    float ev[S_];
    for (int s = 0; s < S_; ++s) {
      ev[s] = __expf(logitsS[m * S_ + s] - mx);
      sm += ev[s];
    }
    float inv = 1.f / sm;
    for (int s = 0; s < S_; ++s) attnS[m * S_ + s] = ev[s] * inv;
  }
  __syncthreads();

  // ---- phase C: nv_agg into svaggH[:,128:256] ----
  for (int e = tid; e < 16 * 128; e += 128) {
    int m = e >> 7, d = e & 127;
    float acc = 0.f;
#pragma unroll
    for (int s = 0; s < S_; ++s)
      acc += attnS[m * S_ + s] * (float)nvH[(m * S_ + s) * 128 + d];
    svaggH[m * 256 + 128 + d] = (_Float16)acc;
  }
  __syncthreads();

  // ---- phase D: out = relu([sv | nv_agg] @ gw3) ----
  for (int nt = wave; nt < 8; nt += 4) {
    v8f acc = {};
#pragma unroll
    for (int kt = 0; kt < 8; ++kt) {  // K = 256
      const int k0 = kt * 32;
      v16h af = cat8(*(const v8h*)&svaggH[ln * 256 + k0 + 16 * hi],
                     *(const v8h*)&svaggH[ln * 256 + k0 + 16 * hi + 8]);
      v16h bf = *(const v16h*)&gw3s[((kt * 8 + nt) * 32 + lane) * 16];
      acc = wmma_f16(af, bf, acc);
    }
    int c = nt * 16 + ln;
#pragma unroll
    for (int i = 0; i < 8; ++i) {
      int r = i + hi * 8;
      if (r < mcount)
        out[((long)b * M + m0 + r) * D_ + c] = fmaxf(acc[i], 0.f);
    }
  }
}

// ---------------------------------------------------------------------------
// output = h_local + s_global ; second output = s_global
// ---------------------------------------------------------------------------
__global__ void final_kernel(const float* __restrict__ hl,
                             const float* __restrict__ sg,
                             float* __restrict__ out) {
  int e = blockIdx.x * blockDim.x + threadIdx.x;
  if (e >= B_ * L_ * D_) return;
  out[e] = hl[e] + sg[e];
  out[B_ * L_ * D_ + e] = sg[e];
}

// ---------------------------------------------------------------------------
extern "C" void kernel_launch(void* const* d_in, const int* in_sizes, int n_in,
                              void* d_out, int out_size, void* d_ws,
                              size_t ws_size, hipStream_t stream) {
  const int*   inputs  = (const int*)d_in[0];
  const int*   adj     = (const int*)d_in[1];
  const int*   mask    = (const int*)d_in[2];
  const int*   item    = (const int*)d_in[3];
  const float* emb     = (const float*)d_in[4];
  const int*   adj_all = (const int*)d_in[5];
  const float* num_w   = (const float*)d_in[6];
  const float* a0      = (const float*)d_in[7];
  const float* a1      = (const float*)d_in[8];
  const float* a2      = (const float*)d_in[9];
  const float* a3      = (const float*)d_in[10];
  const float* gw1     = (const float*)d_in[11];  // [2,129,128]
  const float* gw2     = (const float*)d_in[12];  // [2,128]
  const float* gw3     = (const float*)d_in[13];  // [2,256,128]

  float* ws = (float*)d_ws;
  size_t off = 0;
  float* h    = ws + off; off += (size_t)B_ * L_ * D_;        // entity[0]
  float* hl   = ws + off; off += (size_t)B_ * L_ * D_;        // h_local
  float* sumI = ws + off; off += (size_t)B_ * D_;
  float* outA = ws + off; off += (size_t)B_ * L_ * D_;        // hop0, hp0
  float* outB = ws + off; off += (size_t)B_ * L_ * S_ * D_;   // hop0, hp1
  float* sG   = ws + off; off += (size_t)B_ * L_ * D_;        // s_global
  float* w1   = ws + off; off += (size_t)B_ * L_ * S_;
  float* w2   = ws + off; off += (size_t)B_ * L_ * S_ * S_;
  int* idx1 = (int*)(ws + off); off += (size_t)B_ * L_ * S_;
  int* idx2 = (int*)(ws + off); off += (size_t)B_ * L_ * S_ * S_;
  // fragment-major f16 weights (allocated as float-sized slots, 32B aligned)
  _Float16* gw1s = (_Float16*)(ws + off); off += 2 * 4 * 8 * 32 * 16 / 2;
  _Float16* gw3s = (_Float16*)(ws + off); off += 2 * 8 * 8 * 32 * 16 / 2;

  // ---- weight pre-swizzle (K=128 part of gw1; all of gw3), both hops ----
  {
    int n1 = 4 * 8 * 32 * 16;  // KT=4
    swz_kernel<<<(n1 + 255) / 256, 256, 0, stream>>>(gw1, gw1s, 4);
    swz_kernel<<<(n1 + 255) / 256, 256, 0, stream>>>(gw1 + 129 * 128,
                                                     gw1s + n1, 4);
    int n3 = 8 * 8 * 32 * 16;  // KT=8
    swz_kernel<<<(n3 + 255) / 256, 256, 0, stream>>>(gw3, gw3s, 8);
    swz_kernel<<<(n3 + 255) / 256, 256, 0, stream>>>(gw3 + 256 * 128,
                                                     gw3s + n3, 8);
  }

  local_agg_kernel<<<B_, 128, 0, stream>>>(inputs, adj, emb, a0, a1, a2, a3, h,
                                           hl);
  sum_item_kernel<<<B_, 128, 0, stream>>>(item, mask, emb, sumI);
  {
    int n = B_ * L_ * S_;
    nbr1_kernel<<<(n + 255) / 256, 256, 0, stream>>>(inputs, adj_all, num_w,
                                                     idx1, w1);
  }
  {
    int n = B_ * L_ * S_ * S_;
    nbr2_kernel<<<(n + 255) / 256, 256, 0, stream>>>(idx1, adj_all, num_w,
                                                     idx2, w2);
  }

  const float* G1L_0 = gw1 + 128 * 128;            // nw row, hop 0
  const float* G1L_1 = gw1 + 129 * 128 + 128 * 128;
  const float* G2_0 = gw2;                         const float* G2_1 = gw2 + 128;
  _Float16* G1S_0 = gw1s;                          _Float16* G1S_1 = gw1s + 4 * 8 * 32 * 16;
  _Float16* G3S_0 = gw3s;                          _Float16* G3S_1 = gw3s + 8 * 8 * 32 * 16;

  const size_t dynLds = (2 * 192 * 128 + 32 * 32 * 16) * sizeof(_Float16);  // 128 KB

  // n_hop=0, hp=0: M=50, sv = h (dense), nv = gather(emb, idx1), nw = w1
  {
    dim3 g((L_ + 15) / 16, B_);
    global_agg_kernel<<<g, 128, dynLds, stream>>>(
        emb, sumI, h, nullptr, 0, nullptr, idx1, 1, w1, G1S_0, G1L_0, G2_0,
        G3S_0, outA, L_);
  }
  // n_hop=0, hp=1: M=600, sv = gather(idx1), nv = gather(idx2), nw = w2
  {
    int M = L_ * S_;
    dim3 g((M + 15) / 16, B_);
    global_agg_kernel<<<g, 128, dynLds, stream>>>(
        emb, sumI, nullptr, idx1, 1, nullptr, idx2, 1, w2, G1S_0, G1L_0, G2_0,
        G3S_0, outB, M);
  }
  // n_hop=1, hp=0: M=50, sv = outA (dense), nv = outB (dense), nw = w1
  {
    dim3 g((L_ + 15) / 16, B_);
    global_agg_kernel<<<g, 128, dynLds, stream>>>(
        emb, sumI, outA, nullptr, 0, outB, nullptr, 0, w1, G1S_1, G1L_1, G2_1,
        G3S_1, sG, L_);
  }

  final_kernel<<<(B_ * L_ * D_ + 255) / 256, 256, 0, stream>>>(hl, sG,
                                                               (float*)d_out);
}
